// KB_Mapping_19361712570541
// MI455X (gfx1250) — compile-verified, matrix operand order checked by reading, not run on hardware
//
#include <hip/hip_runtime.h>
#include <stdint.h>

// Problem constants (from reference)
#define CCH   128          // channels
#define NROWS 131072       // sequence length
#define TR    128          // output rows per workgroup
#define ABLK  10           // row-blocks computed in stages A-C (160 rows incl. halo)
#define XR    162          // x rows staged in LDS (160 + conv halo)
#define STR   136          // padded LDS stride in bf16 elems (128 + 8) -> 272B rows, 16B aligned
#define WFSTR 264          // padded stride for fusion weight (256 + 8)
#define HBITS 8388608      // C*N/2 : threefry half-count (fits int32)

typedef __bf16 bf16_t;
typedef __attribute__((ext_vector_type(16))) __bf16 v16bf;
typedef __attribute__((ext_vector_type(8)))  __bf16 v8bf;
typedef __attribute__((ext_vector_type(4)))  __bf16 v4bf;
typedef __attribute__((ext_vector_type(8)))  float  v8f;

// LDS region byte offsets (all 16B aligned)
#define OFF_W1   0
#define OFF_W2   34816
#define OFF_W3   69632
#define OFF_WF   104448
#define OFF_T1   172032
#define OFF_T2   173568
#define OFF_X    175104
#define OFF_BDW  219168     // b1_dw, later reused for b2_dw
#define OFF_B2   262688     // b2,    later reused for b2_pw
#define LDS_TOTAL 306208

// ---------------------------------------------------------------------------
// threefry2x32 with jax PRNGKey(42) => key = (0, 42); constants folded.
// ---------------------------------------------------------------------------
__device__ __forceinline__ void tf2x32(uint32_t x0, uint32_t x1,
                                       uint32_t& y0, uint32_t& y1) {
  const uint32_t k0 = 0u, k1 = 42u, k2 = 0x1BD11BDAu ^ 42u;
  x0 += k0; x1 += k1;
#define TFR(r) { x0 += x1; x1 = (x1 << (r)) | (x1 >> (32 - (r))); x1 ^= x0; }
  TFR(13) TFR(15) TFR(26) TFR(6)   x0 += k1; x1 += k2 + 1u;
  TFR(17) TFR(29) TFR(16) TFR(24)  x0 += k2; x1 += k0 + 2u;
  TFR(13) TFR(15) TFR(26) TFR(6)   x0 += k0; x1 += k1 + 3u;
  TFR(17) TFR(29) TFR(16) TFR(24)  x0 += k1; x1 += k2 + 4u;
  TFR(13) TFR(15) TFR(26) TFR(6)   x0 += k2; x1 += k0 + 5u;
#undef TFR
  y0 = x0; y1 = x1;
}

// uniform(bits) < 0.5  <=>  bits < 2^31  <=>  MSB == 0
__device__ __forceinline__ uint32_t keep_bit(uint32_t u) { return (~u) >> 31; }

// Pre-pass: pack the Bernoulli mask (flat index = c*N + n) into bits.
// Thread t covers threefry counters j = 16t..16t+15; pair is (j, HBITS+j).
__global__ __launch_bounds__(256) void mask_kernel(unsigned short* __restrict__ bits) {
  uint32_t t = blockIdx.x * 256u + threadIdx.x;     // t < HBITS/16
  uint32_t b0 = 0, b1 = 0;
  #pragma unroll
  for (int s = 0; s < 16; ++s) {
    uint32_t j = t * 16u + (uint32_t)s;
    uint32_t y0, y1;
    tf2x32(j, j + (uint32_t)HBITS, y0, y1);
    b0 |= keep_bit(y0) << s;
    b1 |= keep_bit(y1) << s;
  }
  bits[t] = (unsigned short)b0;
  bits[(uint32_t)HBITS / 16u + t] = (unsigned short)b1;
}

// ---------------------------------------------------------------------------
// Fragment loader: 16-bit A/B operand layout per CDNA5 ISA 7.12.2 —
// lane (m = lane&15, hi = lane>>4) holds K = 8*hi + {0..7} and 8*hi + {16..23}:
// two contiguous 16-byte chunks from a row-major [rows][stride] array.
// ---------------------------------------------------------------------------
__device__ __forceinline__ v16bf load_frag(const bf16_t* base, int stride,
                                           int row0, int koff, int lane) {
  int m = lane & 15, hi = lane >> 4;
  const bf16_t* p = base + (row0 + m) * stride + koff + hi * 8;
  v8bf lo = *(const v8bf*)(p);
  v8bf hh = *(const v8bf*)(p + 16);
  return __builtin_shufflevector(lo, hh, 0,1,2,3,4,5,6,7,8,9,10,11,12,13,14,15);
}

__device__ __forceinline__ v4bf cvt4(float4 f) {
  v4bf r;
  r[0] = (bf16_t)f.x; r[1] = (bf16_t)f.y; r[2] = (bf16_t)f.z; r[3] = (bf16_t)f.w;
  return r;
}

#define WMMA_BF16(A, B, Cacc) \
  __builtin_amdgcn_wmma_f32_16x16x32_bf16(false, (A), false, (B), (short)0, (Cacc), false, false)

// ---------------------------------------------------------------------------
// Fused kernel: one workgroup (256 threads = 8 waves) per 128 output rows.
// USE_MASK=true  -> read packed mask bits from global buffer
// USE_MASK=false -> compute threefry inline (fallback when ws too small)
// ---------------------------------------------------------------------------
template <bool USE_MASK>
__global__ __launch_bounds__(256) void fused_kernel(
    const float* __restrict__ xg,
    const float* __restrict__ w_b1_dw,
    const float* __restrict__ w_b1_pw,
    const float* __restrict__ w_b2_1x1,
    const float* __restrict__ w_b2_dw,
    const float* __restrict__ w_b2_pw,
    const float* __restrict__ w_fusion,
    const unsigned char* __restrict__ maskbits,
    float* __restrict__ outg)
{
  extern __shared__ char smem[];
  bf16_t* w1  = (bf16_t*)(smem + OFF_W1);
  bf16_t* w2  = (bf16_t*)(smem + OFF_W2);
  bf16_t* w3  = (bf16_t*)(smem + OFF_W3);
  bf16_t* wfl = (bf16_t*)(smem + OFF_WF);
  float*  t1  = (float*) (smem + OFF_T1);
  float*  t2  = (float*) (smem + OFF_T2);
  bf16_t* xs  = (bf16_t*)(smem + OFF_X);
  bf16_t* bdw = (bf16_t*)(smem + OFF_BDW);
  bf16_t* b2s = (bf16_t*)(smem + OFF_B2);

  const int tid  = threadIdx.x;
  const int lane = tid & 31;
  const int wave = tid >> 5;
  const int m    = lane & 15;
  const int hi   = lane >> 4;
  const int r0   = blockIdx.x * TR;

  // ---- Phase 0: stage weights (f32 -> bf16), taps, x tile into LDS (x4 vectorized) ----
  {
    const float4* pw1 = (const float4*)w_b1_pw;
    const float4* pw2 = (const float4*)w_b2_1x1;
    const float4* pw3 = (const float4*)w_b2_pw;
    for (int idx = tid; idx < CCH * CCH / 4; idx += 256) {
      int o = idx >> 5, i4 = (idx & 31) * 4;
      *(v4bf*)&w1[o * STR + i4] = cvt4(pw1[idx]);
      *(v4bf*)&w2[o * STR + i4] = cvt4(pw2[idx]);
      *(v4bf*)&w3[o * STR + i4] = cvt4(pw3[idx]);
    }
    const float4* pwf = (const float4*)w_fusion;
    for (int idx = tid; idx < CCH * 2 * CCH / 4; idx += 256) {
      int o = idx >> 6, i4 = (idx & 63) * 4;
      *(v4bf*)&wfl[o * WFSTR + i4] = cvt4(pwf[idx]);
    }
    for (int idx = tid; idx < 3 * CCH; idx += 256) {
      int kh = idx >> 7, c = idx & 127;     // only middle column (kw=1) matters: W=1 image
      t1[idx] = w_b1_dw[c * 9 + kh * 3 + 1];
      t2[idx] = w_b2_dw[c * 9 + kh * 3 + 1];
    }
    const float4* px = (const float4*)xg;
    for (int idx = tid; idx < XR * (CCH / 4); idx += 256) {
      int rr = idx >> 5, c4 = (idx & 31) * 4;
      int gr = r0 - 17 + rr;                // x LDS row 0 <-> global row r0-17
      float4 f = make_float4(0.f, 0.f, 0.f, 0.f);
      if (gr >= 0 && gr < NROWS) f = px[gr * (CCH / 4) + (c4 >> 2)];
      *(v4bf*)&xs[rr * STR + c4] = cvt4(f);
    }
  }
  __syncthreads();

  // ---- Phase 1: b1_dw = relu(3-tap conv over rows), 160 rows [r0-16, r0+144) ----
  for (int idx = tid; idx < ABLK * 16 * (CCH / 4); idx += 256) {
    int rr = idx >> 5, c4 = (idx & 31) * 4;
    v4bf a = *(const v4bf*)&xs[rr * STR + c4];
    v4bf b = *(const v4bf*)&xs[(rr + 1) * STR + c4];
    v4bf d = *(const v4bf*)&xs[(rr + 2) * STR + c4];
    const float4 ta = *(const float4*)&t1[c4];
    const float4 tb = *(const float4*)&t1[128 + c4];
    const float4 tc = *(const float4*)&t1[256 + c4];
    v4bf r;
    r[0] = (bf16_t)fmaxf((float)a[0]*ta.x + (float)b[0]*tb.x + (float)d[0]*tc.x, 0.f);
    r[1] = (bf16_t)fmaxf((float)a[1]*ta.y + (float)b[1]*tb.y + (float)d[1]*tc.y, 0.f);
    r[2] = (bf16_t)fmaxf((float)a[2]*ta.z + (float)b[2]*tb.z + (float)d[2]*tc.z, 0.f);
    r[3] = (bf16_t)fmaxf((float)a[3]*ta.w + (float)b[3]*tb.w + (float)d[3]*tc.w, 0.f);
    *(v4bf*)&bdw[rr * STR + c4] = r;
  }
  __syncthreads();

  // ---- Phase 2: b1 = relu(b1_dw @ W1ᵀ); b2 = (relu(x @ W2ᵀ) + b1) * mask ----
  // Wave owns output column-block cb = wave; B fragments (W1, W2 columns) hoisted.
  {
    const int cb = wave;
    v16bf bw1[4], bw2[4];
    #pragma unroll
    for (int kb = 0; kb < 4; ++kb) {
      bw1[kb] = load_frag(w1, STR, cb * 16, kb * 32, lane);
      bw2[kb] = load_frag(w2, STR, cb * 16, kb * 32, lane);
    }
    const int ch = cb * 16 + m;
    const int bitcol = ch * NROWS;                 // fits int32 (<= 2^24)
    for (int rb = 0; rb < ABLK; ++rb) {
      const int rowbase = r0 - 16 + rb * 16 + hi * 8;  // multiple of 8
      // Issue mask byte load first; 8 WMMAs below hide its latency.
      unsigned int mb = 0xFFu;
      if (USE_MASK) {
        int bidx = (bitcol + rowbase) >> 3;        // 32-bit arithmetic shift
        mb = maskbits[bidx < 0 ? 0 : bidx];        // halo rows are zero anyway
      }
      v8f acc1 = {0.f,0.f,0.f,0.f,0.f,0.f,0.f,0.f};
      v8f acc2 = {0.f,0.f,0.f,0.f,0.f,0.f,0.f,0.f};
      #pragma unroll
      for (int kb = 0; kb < 4; ++kb) {
        v16bf a1 = load_frag(bdw, STR, rb * 16, kb * 32, lane);
        acc1 = WMMA_BF16(a1, bw1[kb], acc1);
      }
      #pragma unroll
      for (int kb = 0; kb < 4; ++kb) {
        v16bf a2 = load_frag(xs, STR, rb * 16 + 1, kb * 32, lane);
        acc2 = WMMA_BF16(a2, bw2[kb], acc2);
      }
      #pragma unroll
      for (int v = 0; v < 8; ++v) {
        float s = fmaxf(acc1[v], 0.0f) + fmaxf(acc2[v], 0.0f);
        uint32_t keep;
        if (USE_MASK) {
          keep = (mb >> v) & 1u;
        } else {
          int flat = bitcol + rowbase + v;
          flat = flat < 0 ? 0 : flat;
          uint32_t sec = (flat >= HBITS) ? 1u : 0u;
          uint32_t jj  = (uint32_t)flat - (sec ? (uint32_t)HBITS : 0u);
          uint32_t y0, y1; tf2x32(jj, jj + (uint32_t)HBITS, y0, y1);
          keep = keep_bit(sec ? y1 : y0);
        }
        b2s[(rb * 16 + v + hi * 8) * STR + ch] = (bf16_t)(keep ? s : 0.0f);
      }
    }
  }
  __syncthreads();

  // ---- Phase 3: b2_dw = relu(3-tap conv on b2) for the 128 output rows ----
  // b2 LDS row 0 <-> global r0-16, so output row rr uses b2 rows rr+15..rr+17.
  for (int idx = tid; idx < TR * (CCH / 4); idx += 256) {
    int rr = idx >> 5, c4 = (idx & 31) * 4;
    v4bf a = *(const v4bf*)&b2s[(rr + 15) * STR + c4];
    v4bf b = *(const v4bf*)&b2s[(rr + 16) * STR + c4];
    v4bf d = *(const v4bf*)&b2s[(rr + 17) * STR + c4];
    const float4 ta = *(const float4*)&t2[c4];
    const float4 tb = *(const float4*)&t2[128 + c4];
    const float4 tc = *(const float4*)&t2[256 + c4];
    v4bf r;
    r[0] = (bf16_t)fmaxf((float)a[0]*ta.x + (float)b[0]*tb.x + (float)d[0]*tc.x, 0.f);
    r[1] = (bf16_t)fmaxf((float)a[1]*ta.y + (float)b[1]*tb.y + (float)d[1]*tc.y, 0.f);
    r[2] = (bf16_t)fmaxf((float)a[2]*ta.z + (float)b[2]*tb.z + (float)d[2]*tc.z, 0.f);
    r[3] = (bf16_t)fmaxf((float)a[3]*ta.w + (float)b[3]*tb.w + (float)d[3]*tc.w, 0.f);
    *(v4bf*)&bdw[rr * STR + c4] = r;     // reuse b1_dw region
  }
  __syncthreads();

  // ---- Phase 4: b2_pw = relu(b2_dw @ W3ᵀ); wave owns row-block rb = wave ----
  {
    const int rb = wave;
    v16bf a3[4];
    #pragma unroll
    for (int kb = 0; kb < 4; ++kb) a3[kb] = load_frag(bdw, STR, rb * 16, kb * 32, lane);
    for (int ocb = 0; ocb < 8; ++ocb) {
      v8f acc = {0.f,0.f,0.f,0.f,0.f,0.f,0.f,0.f};
      #pragma unroll
      for (int kb = 0; kb < 4; ++kb) {
        v16bf b = load_frag(w3, STR, ocb * 16, kb * 32, lane);
        acc = WMMA_BF16(a3[kb], b, acc);
      }
      #pragma unroll
      for (int v = 0; v < 8; ++v)   // reuse b2 region for b2_pw (own rows only)
        b2s[(rb * 16 + v + hi * 8) * STR + ocb * 16 + m] = (bf16_t)fmaxf(acc[v], 0.0f);
    }

    // ---- Phase 5: out = relu([x ; b2_pw] @ Wfᵀ)  (K = 256) ----
    // Same-wave LDS RAW on b2_pw: DScnt ordering, compiler inserts waits.
    v16bf axf[4], apf[4];
    #pragma unroll
    for (int kb = 0; kb < 4; ++kb) {
      axf[kb] = load_frag(xs,  STR, rb * 16 + 17, kb * 32, lane);
      apf[kb] = load_frag(b2s, STR, rb * 16,      kb * 32, lane);
    }
    for (int ocb = 0; ocb < 8; ++ocb) {
      v8f acc = {0.f,0.f,0.f,0.f,0.f,0.f,0.f,0.f};
      #pragma unroll
      for (int kb = 0; kb < 4; ++kb) {
        v16bf b = load_frag(wfl, WFSTR, ocb * 16, kb * 32, lane);        // K 0..127: x half
        acc = WMMA_BF16(axf[kb], b, acc);
      }
      #pragma unroll
      for (int kb = 0; kb < 4; ++kb) {
        v16bf b = load_frag(wfl, WFSTR, ocb * 16, 128 + kb * 32, lane);  // K 128..255: b2_pw
        acc = WMMA_BF16(apf[kb], b, acc);
      }
      #pragma unroll
      for (int v = 0; v < 8; ++v) {
        int row = r0 + rb * 16 + v + hi * 8;
        outg[row * CCH + ocb * 16 + m] = fmaxf(acc[v], 0.0f);
      }
    }
  }
}

// ---------------------------------------------------------------------------
extern "C" void kernel_launch(void* const* d_in, const int* in_sizes, int n_in,
                              void* d_out, int out_size, void* d_ws, size_t ws_size,
                              hipStream_t stream) {
  (void)in_sizes; (void)n_in; (void)out_size;
  const float* xg       = (const float*)d_in[0];
  const float* w_b1_dw  = (const float*)d_in[1];
  const float* w_b1_pw  = (const float*)d_in[2];
  const float* w_b2_1x1 = (const float*)d_in[3];
  const float* w_b2_dw  = (const float*)d_in[4];
  const float* w_b2_pw  = (const float*)d_in[5];
  const float* w_fusion = (const float*)d_in[6];

  const size_t mask_bytes = (size_t)CCH * NROWS / 8;   // 2 MB packed bits
  dim3 grid(NROWS / TR);                               // 1024 workgroups

  if (d_ws != nullptr && ws_size >= mask_bytes) {
    dim3 mg(HBITS / 16 / 256);                         // 2048 blocks x 256 threads
    mask_kernel<<<mg, 256, 0, stream>>>((unsigned short*)d_ws);
    fused_kernel<true><<<grid, 256, LDS_TOTAL, stream>>>(
        xg, w_b1_dw, w_b1_pw, w_b2_1x1, w_b2_dw, w_b2_pw, w_fusion,
        (const unsigned char*)d_ws, (float*)d_out);
  } else {
    fused_kernel<false><<<grid, 256, LDS_TOTAL, stream>>>(
        xg, w_b1_dw, w_b1_pw, w_b2_1x1, w_b2_dw, w_b2_pw, w_fusion,
        nullptr, (float*)d_out);
  }
}
